// AdvancedTransformer_50964081934910
// MI455X (gfx1250) — matrix operand order repaired and checked by reading, not compiled, not run on hardware
//
#include <hip/hip_runtime.h>

typedef unsigned short u16;
typedef __attribute__((ext_vector_type(16))) __bf16 v16bf;
typedef __attribute__((ext_vector_type(8)))  float  v8f;

#define S_LEN   2048
#define DMODEL  1024
#define NHEAD   16
#define DK_HEAD 64
#define DFF     4096
#define NLAYERS 4

#define BK   32
#define SSTR 40   // LDS row stride in halves (BK + 8 pad), keeps b64/b32 alignment

__device__ __forceinline__ u16 f2bf(float f) {
  unsigned u = __float_as_uint(f);
  unsigned r = (u + 0x7fffu + ((u >> 16) & 1u)) >> 16;  // round-to-nearest-even
  return (u16)r;
}

__device__ __forceinline__ float gelu_tanh(float v) {
  float c = 0.7978845608028654f * (v + 0.044715f * v * v * v);
  return 0.5f * v * (1.0f + tanhf(c));
}

// ---------------------------------------------------------------------------
// Tiled bf16 WMMA GEMM, double-buffered LDS, optional split-K over blockIdx.z.
//   C[M,N] = A[M,K] * B (+bias, optional GELU).  A row-major bf16 (lda).
//   opB_T=0: B is [K,N] row-major; opB_T=1: B is [N,K] row-major (A*B^T).
//   gridDim.z > 1: K split into gridDim.z chunks; chunk z writes partials to
//   C + z*cStrideZ (no bias/act in that mode).
// Wave layout: 8 waves = WMW x WNW, each wave owns (MIt*16) x (NIt*16).
// ---------------------------------------------------------------------------
template <int BMt, int BNt, int WMW, int WNW>
__global__ __launch_bounds__(256) void k_gemm_bf16(
    const u16* __restrict__ A, int lda,
    const u16* __restrict__ B, int ldb,
    float* __restrict__ C, int ldc, long cStrideZ,
    int M, int N, int K,
    const float* __restrict__ bias, int opB_T, int act_gelu)
{
  constexpr int MIt = BMt / (WMW * 16);
  constexpr int NIt = BNt / (WNW * 16);
  constexpr int ACH_TOT = BMt * (BK / 8);           // 8-half chunks in A tile
  constexpr int ACH = (ACH_TOT + 255) / 256;
  constexpr int BT_TOT = BNt * (BK / 8);            // opT chunks in B tile
  constexpr int BTCH = (BT_TOT + 255) / 256;
  constexpr int BN_TOT = (BK / 2) * (BNt / 8);      // opN double-row chunks
  constexpr int BNCH = (BN_TOT + 255) / 256;
  constexpr int RBN = (BTCH > 2 * BNCH) ? BTCH : 2 * BNCH;

  __shared__ u16 sA[2 * BMt * SSTR];
  __shared__ u16 sB[2 * BNt * SSTR];   // stored as [n][k]

  const int tid  = threadIdx.x;
  const int lane = tid & 31;
  const int wid  = tid >> 5;
  const int wm   = wid % WMW;
  const int wn   = wid / WMW;
  const int bm0  = blockIdx.y * BMt;
  const int bn0  = blockIdx.x * BNt;
  const int lh   = lane & 15;
  const int hi   = lane >> 4;
  const int kbA  = hi * 8;     // A frag: K 0-7/16-23 vs 8-15/24-31
  const int kbB  = hi * 8;     // B frag: K 0-7/16-23 vs 8-15/24-31

  const int kLen  = K / (int)gridDim.z;
  const int kBase = (int)blockIdx.z * kLen;
  const u16* Ap = A + kBase;
  const u16* Bp = opB_T ? (B + kBase) : (B + (size_t)kBase * ldb);
  float* Cp = C + (size_t)blockIdx.z * cStrideZ;

  const uint4 z4 = make_uint4(0u, 0u, 0u, 0u);

  auto loadA = [&](int k0, uint4* r) {
#pragma unroll
    for (int i = 0; i < ACH; ++i) {
      int c = tid + i * 256;
      uint4 d = z4;
      if (c < ACH_TOT) {
        int rr = c >> 2, cc = (c & 3) << 3;
        int gr = bm0 + rr;
        if (gr < M) {
          const u16* gp = Ap + (size_t)gr * lda + (k0 + cc);
          d = *(const uint4*)gp;
          if (k0 + BK < kLen) __builtin_prefetch(gp + BK, 0, 3);
        }
      }
      r[i] = d;
    }
  };
  auto storeA = [&](int buf, const uint4* r) {
    u16* base = sA + buf * (BMt * SSTR);
#pragma unroll
    for (int i = 0; i < ACH; ++i) {
      int c = tid + i * 256;
      if (c >= ACH_TOT) continue;
      int rr = c >> 2, cc = (c & 3) << 3;
      uint2 lo;  lo.x = r[i].x;  lo.y = r[i].y;
      uint2 hi2; hi2.x = r[i].z; hi2.y = r[i].w;
      *(uint2*)(base + rr * SSTR + cc)     = lo;
      *(uint2*)(base + rr * SSTR + cc + 4) = hi2;
    }
  };
  auto loadBT = [&](int k0, uint4* r) {          // B [N x K] row-major
#pragma unroll
    for (int i = 0; i < BTCH; ++i) {
      int c = tid + i * 256;
      uint4 d = z4;
      if (c < BT_TOT) {
        int rr = c >> 2, cc = (c & 3) << 3;
        int gn = bn0 + rr;
        if (gn < N) {
          const u16* gp = Bp + (size_t)gn * ldb + (k0 + cc);
          d = *(const uint4*)gp;
          if (k0 + BK < kLen) __builtin_prefetch(gp + BK, 0, 3);
        }
      }
      r[i] = d;
    }
  };
  auto storeBT = [&](int buf, const uint4* r) {
    u16* base = sB + buf * (BNt * SSTR);
#pragma unroll
    for (int i = 0; i < BTCH; ++i) {
      int c = tid + i * 256;
      if (c >= BT_TOT) continue;
      int rr = c >> 2, cc = (c & 3) << 3;
      uint2 lo;  lo.x = r[i].x;  lo.y = r[i].y;
      uint2 hi2; hi2.x = r[i].z; hi2.y = r[i].w;
      *(uint2*)(base + rr * SSTR + cc)     = lo;
      *(uint2*)(base + rr * SSTR + cc + 4) = hi2;
    }
  };
  auto loadBN = [&](int k0, uint4* r) {          // B [K x N] row-major, 2 k-rows
#pragma unroll
    for (int i = 0; i < BNCH; ++i) {
      int c = tid + i * 256;
      uint4 d0 = z4, d1 = z4;
      if (c < BN_TOT) {
        int kr = (c / (BNt / 8)) * 2;
        int nc = (c % (BNt / 8)) * 8;
        int gn = bn0 + nc;
        if (gn < N) {
          const u16* gp = Bp + (size_t)(k0 + kr) * ldb + gn;
          d0 = *(const uint4*)gp;
          d1 = *(const uint4*)(gp + ldb);
          if (k0 + BK < kLen) __builtin_prefetch(gp + (size_t)BK * ldb, 0, 3);
        }
      }
      r[2 * i] = d0; r[2 * i + 1] = d1;
    }
  };
  auto storeBN = [&](int buf, const uint4* r) {  // transpose: packed k-pairs, b32
    u16* base = sB + buf * (BNt * SSTR);
#pragma unroll
    for (int i = 0; i < BNCH; ++i) {
      int c = tid + i * 256;
      if (c >= BN_TOT) continue;
      int kr = (c / (BNt / 8)) * 2;
      int nc = (c % (BNt / 8)) * 8;
      const u16* h0 = (const u16*)&r[2 * i];
      const u16* h1 = (const u16*)&r[2 * i + 1];
#pragma unroll
      for (int j = 0; j < 8; ++j) {
        unsigned pk = (unsigned)h0[j] | ((unsigned)h1[j] << 16);
        *(unsigned*)(base + (nc + j) * SSTR + kr) = pk;
      }
    }
  };

  v8f acc[MIt][NIt];
#pragma unroll
  for (int mi = 0; mi < MIt; ++mi)
#pragma unroll
    for (int ni = 0; ni < NIt; ++ni)
#pragma unroll
      for (int e = 0; e < 8; ++e) acc[mi][ni][e] = 0.0f;

  union Frag { v16bf v; uint2 u[4]; };

  uint4 ra[ACH], rb[RBN];
  loadA(0, ra);
  if (opB_T) loadBT(0, rb); else loadBN(0, rb);

  int cur = 0;
  for (int k0 = 0; k0 < kLen; k0 += BK) {
    storeA(cur, ra);
    if (opB_T) storeBT(cur, rb); else storeBN(cur, rb);
    __syncthreads();
    if (k0 + BK < kLen) {
      loadA(k0 + BK, ra);
      if (opB_T) loadBT(k0 + BK, rb); else loadBN(k0 + BK, rb);
    }

    const u16* baseA = sA + cur * (BMt * SSTR);
    const u16* baseB = sB + cur * (BNt * SSTR);
    Frag fa[MIt], fb[NIt];
#pragma unroll
    for (int mi = 0; mi < MIt; ++mi) {
      const u16* p = baseA + ((wm * MIt + mi) * 16 + lh) * SSTR + kbA;
      fa[mi].u[0] = *(const uint2*)(p);
      fa[mi].u[1] = *(const uint2*)(p + 4);
      fa[mi].u[2] = *(const uint2*)(p + 16);
      fa[mi].u[3] = *(const uint2*)(p + 20);
    }
#pragma unroll
    for (int ni = 0; ni < NIt; ++ni) {
      const u16* p = baseB + ((wn * NIt + ni) * 16 + lh) * SSTR + kbB;
      fb[ni].u[0] = *(const uint2*)(p);
      fb[ni].u[1] = *(const uint2*)(p + 4);
      fb[ni].u[2] = *(const uint2*)(p + 16);
      fb[ni].u[3] = *(const uint2*)(p + 20);
    }
#pragma unroll
    for (int mi = 0; mi < MIt; ++mi)
#pragma unroll
      for (int ni = 0; ni < NIt; ++ni)
        acc[mi][ni] = __builtin_amdgcn_wmma_f32_16x16x32_bf16(
            false, fa[mi].v, false, fb[ni].v, (short)0, acc[mi][ni], false, false);
    cur ^= 1;
  }

  // epilogue
#pragma unroll
  for (int mi = 0; mi < MIt; ++mi) {
#pragma unroll
    for (int ni = 0; ni < NIt; ++ni) {
      int gn = bn0 + (wn * NIt + ni) * 16 + lh;
      if (gn >= N) continue;
      float bv = bias ? bias[gn] : 0.0f;
      int gm0 = bm0 + (wm * MIt + mi) * 16 + hi * 8;
#pragma unroll
      for (int r = 0; r < 8; ++r) {
        int gm = gm0 + r;
        if (gm < M) {
          float v = acc[mi][ni][r] + bv;
          if (act_gelu) v = gelu_tanh(v);
          Cp[(size_t)gm * ldc + gn] = v;
        }
      }
    }
  }
}

// sum nparts partial [rows x cols] buffers into out (leading dim outLd)
__global__ __launch_bounds__(256) void k_reduce(const float* __restrict__ parts,
                                                float* __restrict__ out,
                                                int rows, int cols, int outLd,
                                                int nparts) {
  int idx = blockIdx.x * 256 + threadIdx.x;
  if (idx >= rows * cols) return;
  long pstride = (long)rows * cols;
  float s = 0.0f;
  for (int z = 0; z < nparts; ++z) s += parts[z * pstride + idx];
  int r = idx / cols, c = idx % cols;
  out[(size_t)r * outLd + c] = s;
}

// fp32 -> bf16, 4 elems/thread (n multiple of 1024)
__global__ __launch_bounds__(256) void k_cvt4(const float* __restrict__ s,
                                              u16* __restrict__ d, int n) {
  int i = (blockIdx.x * 256 + threadIdx.x) * 4;
  if (i + 3 < n) {
    float4 f = *(const float4*)(s + i);
    uint2 o;
    o.x = (unsigned)f2bf(f.x) | ((unsigned)f2bf(f.y) << 16);
    o.y = (unsigned)f2bf(f.z) | ((unsigned)f2bf(f.w) << 16);
    *(uint2*)(d + i) = o;
  }
}

__global__ __launch_bounds__(256) void k_copycvt4(const float* __restrict__ s,
                                                  float* __restrict__ d,
                                                  u16* __restrict__ db, int n) {
  int i = (blockIdx.x * 256 + threadIdx.x) * 4;
  if (i + 3 < n) {
    float4 f = *(const float4*)(s + i);
    *(float4*)(d + i) = f;
    uint2 o;
    o.x = (unsigned)f2bf(f.x) | ((unsigned)f2bf(f.y) << 16);
    o.y = (unsigned)f2bf(f.z) | ((unsigned)f2bf(f.w) << 16);
    *(uint2*)(db + i) = o;
  }
}

// row softmax with scale; fp32 in, bf16 out. one block per row, cols<=4096
__global__ __launch_bounds__(256) void k_softmax(const float* __restrict__ S,
                                                 u16* __restrict__ P,
                                                 int cols, float scale) {
  const int row = blockIdx.x;
  const float* sr = S + (size_t)row * cols;
  u16* pr = P + (size_t)row * cols;
  __shared__ float red[8];
  const int tid = threadIdx.x, lane = tid & 31, wid = tid >> 5;
  const int nIt = (cols + 255) >> 8;
  float pv[16];
  float m = -3.4e38f;
  for (int i = 0; i < nIt; ++i) {
    int c = tid + (i << 8);
    if (c < cols) { float v = sr[c] * scale; pv[i] = v; m = fmaxf(m, v); }
  }
  for (int o = 16; o > 0; o >>= 1) m = fmaxf(m, __shfl_xor(m, o, 32));
  if (!lane) red[wid] = m;
  __syncthreads();
  m = red[0];
  for (int i = 1; i < 8; ++i) m = fmaxf(m, red[i]);
  float s = 0.0f;
  for (int i = 0; i < nIt; ++i) {
    int c = tid + (i << 8);
    if (c < cols) { float e = __expf(pv[i] - m); pv[i] = e; s += e; }
  }
  for (int o = 16; o > 0; o >>= 1) s += __shfl_xor(s, o, 32);
  __syncthreads();
  if (!lane) red[wid] = s;
  __syncthreads();
  s = 0.0f;
  for (int i = 0; i < 8; ++i) s += red[i];
  float inv = 1.0f / s;
  for (int i = 0; i < nIt; ++i) {
    int c = tid + (i << 8);
    if (c < cols) pr[c] = f2bf(pv[i] * inv);
  }
}

// x = LayerNorm(x + t)*g + b; writes fp32 x and bf16 shadow. D % 256 == 0
__global__ __launch_bounds__(256) void k_res_ln(float* __restrict__ x,
                                                const float* __restrict__ t,
                                                const float* __restrict__ g,
                                                const float* __restrict__ b,
                                                u16* __restrict__ xb, int D) {
  const int row = blockIdx.x;
  float* xr = x + (size_t)row * D;
  const float* tr = t + (size_t)row * D;
  u16* xbr = xb + (size_t)row * D;
  __shared__ float red[8];
  const int tid = threadIdx.x, lane = tid & 31, wid = tid >> 5;
  const int nIt = D >> 8;
  float v[8];
  float s = 0.0f;
  for (int i = 0; i < nIt; ++i) {
    int c = tid + (i << 8);
    v[i] = xr[c] + tr[c];
    s += v[i];
  }
  for (int o = 16; o > 0; o >>= 1) s += __shfl_xor(s, o, 32);
  if (!lane) red[wid] = s;
  __syncthreads();
  s = 0.0f; for (int i = 0; i < 8; ++i) s += red[i];
  float mu = s / (float)D;
  __syncthreads();
  float q = 0.0f;
  for (int i = 0; i < nIt; ++i) { float d = v[i] - mu; q += d * d; }
  for (int o = 16; o > 0; o >>= 1) q += __shfl_xor(q, o, 32);
  if (!lane) red[wid] = q;
  __syncthreads();
  q = 0.0f; for (int i = 0; i < 8; ++i) q += red[i];
  float rstd = rsqrtf(q / (float)D + 1e-5f);
  for (int i = 0; i < nIt; ++i) {
    int c = tid + (i << 8);
    float o = (v[i] - mu) * rstd * g[c] + b[c];
    xr[c] = o;
    xbr[c] = f2bf(o);
  }
}

static inline void gemm_big(hipStream_t st, const u16* A, int lda, const u16* B,
                            int ldb, float* C, int ldc, int M, int N, int K,
                            const float* bias, int opT, int act) {
  dim3 grid((N + 255) / 256, (M + 127) / 128, 1);
  k_gemm_bf16<128, 256, 2, 4><<<grid, dim3(256), 0, st>>>(
      A, lda, B, ldb, C, ldc, 0L, M, N, K, bias, opT, act);
}
static inline void gemm_ctx_splitk(hipStream_t st, const u16* A, int lda,
                                   const u16* B, int ldb, float* Cparts,
                                   int M, int N, int K, int splitZ) {
  dim3 grid((N + 63) / 64, (M + 127) / 128, splitZ);
  k_gemm_bf16<128, 64, 8, 1><<<grid, dim3(256), 0, st>>>(
      A, lda, B, ldb, Cparts, N, (long)M * N, M, N, K, nullptr, 0, 0);
}

extern "C" void kernel_launch(void* const* d_in, const int* in_sizes, int n_in,
                              void* d_out, int out_size, void* d_ws, size_t ws_size,
                              hipStream_t stream) {
  const float* X    = (const float*)d_in[0];
  const float* WQ   = (const float*)d_in[1];
  const float* BQv  = (const float*)d_in[2];
  const float* WK   = (const float*)d_in[3];
  const float* BKv  = (const float*)d_in[4];
  const float* WV   = (const float*)d_in[5];
  const float* BVv  = (const float*)d_in[6];
  const float* WO   = (const float*)d_in[7];
  const float* BOv  = (const float*)d_in[8];
  const float* W1   = (const float*)d_in[9];
  const float* B1v  = (const float*)d_in[10];
  const float* W2   = (const float*)d_in[11];
  const float* B2v  = (const float*)d_in[12];
  const float* LN1G = (const float*)d_in[13];
  const float* LN1B = (const float*)d_in[14];
  const float* LN2G = (const float*)d_in[15];
  const float* LN2B = (const float*)d_in[16];
  float* xo = (float*)d_out;

  char* ws = (char*)d_ws;
  const size_t MB = 1ull << 20;
  // workspace layout (high-water ~92 MB), aliases noted
  u16*   xb   = (u16*)  (ws + 0 * MB);    // 4 MB  bf16 running x
  u16*   wbuf = (u16*)  (ws + 4 * MB);    // 8 MB  bf16 weight staging
  float* qf   = (float*)(ws + 12 * MB);   // 8 MB
  float* kf   = (float*)(ws + 20 * MB);   // 8 MB
  float* vf   = (float*)(ws + 28 * MB);   // 8 MB
  float* sc   = (float*)(ws + 36 * MB);   // 16 MB scores (per head)
  u16*   pb   = (u16*)  (ws + 52 * MB);   // 8 MB  softmax probs bf16
  float* ctx  = (float*)(ws + 60 * MB);   // 8 MB
  float* tmp  = (float*)(ws + 68 * MB);   // 8 MB  attn_out / ffn_out
  u16*   qb   = (u16*)  (ws + 76 * MB);   // 4 MB
  u16*   kb   = (u16*)  (ws + 80 * MB);   // 4 MB
  u16*   vb   = (u16*)  (ws + 84 * MB);   // 4 MB
  float* ctxp = (float*)(ws + 88 * MB);   // 4 MB  split-K partials (8 x 512 KB)
  float* h1   = (float*)(ws + 12 * MB);   // 32 MB alias (qf/kf/vf/sc dead in FFN)
  u16*   h1b  = (u16*)  (ws + 44 * MB);   // 16 MB alias (sc tail / pb dead in FFN)
  u16*   ctxb = qb;                       // alias (qb dead after attention GEMMs)

  const int SD = S_LEN * DMODEL;   // 2M
  const int SF = S_LEN * DFF;      // 8M
  const int DD = DMODEL * DMODEL;  // 1M
  const int DF = DMODEL * DFF;     // 4M

  k_copycvt4<<<SD / 1024, 256, 0, stream>>>(X, xo, xb, SD);

  for (int l = 0; l < NLAYERS; ++l) {
    const float* wq = WQ + (size_t)l * DD;
    const float* wk = WK + (size_t)l * DD;
    const float* wv = WV + (size_t)l * DD;
    const float* wo = WO + (size_t)l * DD;
    const float* w1 = W1 + (size_t)l * DF;
    const float* w2 = W2 + (size_t)l * DF;
    const float* bq = BQv + (size_t)l * DMODEL;
    const float* bk = BKv + (size_t)l * DMODEL;
    const float* bv = BVv + (size_t)l * DMODEL;
    const float* bo = BOv + (size_t)l * DMODEL;
    const float* b1 = B1v + (size_t)l * DFF;
    const float* b2 = B2v + (size_t)l * DMODEL;

    // QKV projections (bf16 WMMA, fp32 out)
    k_cvt4<<<DD / 1024, 256, 0, stream>>>(wq, wbuf, DD);
    gemm_big(stream, xb, DMODEL, wbuf, DMODEL, qf, DMODEL, S_LEN, DMODEL, DMODEL, bq, 0, 0);
    k_cvt4<<<DD / 1024, 256, 0, stream>>>(wk, wbuf, DD);
    gemm_big(stream, xb, DMODEL, wbuf, DMODEL, kf, DMODEL, S_LEN, DMODEL, DMODEL, bk, 0, 0);
    k_cvt4<<<DD / 1024, 256, 0, stream>>>(wv, wbuf, DD);
    gemm_big(stream, xb, DMODEL, wbuf, DMODEL, vf, DMODEL, S_LEN, DMODEL, DMODEL, bv, 0, 0);
    k_cvt4<<<SD / 1024, 256, 0, stream>>>(qf, qb, SD);
    k_cvt4<<<SD / 1024, 256, 0, stream>>>(kf, kb, SD);
    k_cvt4<<<SD / 1024, 256, 0, stream>>>(vf, vb, SD);

    // attention, head by head
    for (int h = 0; h < NHEAD; ++h) {
      gemm_big(stream, qb + h * DK_HEAD, DMODEL, kb + h * DK_HEAD, DMODEL,
               sc, S_LEN, S_LEN, S_LEN, DK_HEAD, nullptr, /*opB_T=*/1, 0);
      k_softmax<<<S_LEN, 256, 0, stream>>>(sc, pb, S_LEN, 0.125f); // 1/sqrt(64)
      gemm_ctx_splitk(stream, pb, S_LEN, vb + h * DK_HEAD, DMODEL,
                      ctxp, S_LEN, DK_HEAD, S_LEN, /*splitZ=*/8);
      k_reduce<<<(S_LEN * DK_HEAD) / 256, 256, 0, stream>>>(
          ctxp, ctx + h * DK_HEAD, S_LEN, DK_HEAD, DMODEL, 8);
    }

    // output projection + residual LN1
    k_cvt4<<<SD / 1024, 256, 0, stream>>>(ctx, ctxb, SD);
    k_cvt4<<<DD / 1024, 256, 0, stream>>>(wo, wbuf, DD);
    gemm_big(stream, ctxb, DMODEL, wbuf, DMODEL, tmp, DMODEL, S_LEN, DMODEL, DMODEL, bo, 0, 0);
    k_res_ln<<<S_LEN, 256, 0, stream>>>(xo, tmp, LN1G + (size_t)l * DMODEL,
                                        LN1B + (size_t)l * DMODEL, xb, DMODEL);

    // FFN (GELU fused in GEMM epilogue) + residual LN2
    k_cvt4<<<DF / 1024, 256, 0, stream>>>(w1, wbuf, DF);
    gemm_big(stream, xb, DMODEL, wbuf, DFF, h1, DFF, S_LEN, DFF, DMODEL, b1, 0, /*gelu=*/1);
    k_cvt4<<<SF / 1024, 256, 0, stream>>>(h1, h1b, SF);
    k_cvt4<<<DF / 1024, 256, 0, stream>>>(w2, wbuf, DF);
    gemm_big(stream, h1b, DFF, wbuf, DMODEL, tmp, DMODEL, S_LEN, DMODEL, DFF, b2, 0, 0);
    k_res_ln<<<S_LEN, 256, 0, stream>>>(xo, tmp, LN2G + (size_t)l * DMODEL,
                                        LN2B + (size_t)l * DMODEL, xb, DMODEL);
  }
  (void)in_sizes; (void)n_in; (void)out_size; (void)ws_size;
}
